// Mamba_3667902071214
// MI455X (gfx1250) — matrix operand order repaired and checked by reading, not compiled, not run on hardware
//
#include <hip/hip_runtime.h>
#include <hip/hip_bf16.h>

// ---------------------------------------------------------------------------
// Mamba forward (B=1, L=2048, D_MODEL=768, D_IN=1536, N=16, DT_RANK=48,
// DCONV=4, NLAYER=2, VOCAB=32000) for gfx1250 (CDNA5, wave32, WMMA).
//
// GEMMs: split-f16 (v = hi + lo) with f32 accumulation via
// v_wmma_f32_16x16x32_f16.  The hi/lo split is precomputed ONCE per operand
// per launch (separate split kernel), so the GEMM inner loop is pure wide
// LDS loads + WMMA: no f32<->f16 conversion VALU in the hot loop.
// ---------------------------------------------------------------------------

typedef __attribute__((ext_vector_type(16))) _Float16 v16h;
typedef __attribute__((ext_vector_type(8)))  _Float16 v8h;
typedef __attribute__((ext_vector_type(8)))  float    v8f;

#define LSEQ   2048
#define DMODEL 768
#define DIN    1536
#define NST    16
#define DTRANK 48
#define NLAYER 2
#define VOCAB  32000

#define TILE_M 64
#define TILE_N 64
#define TILE_K 32
#define LDSTRIDE 40   // halves; 80 bytes, multiple of 16B, staggers banks

// ---------------------------------------------------------------------------
// f32 -> (hi f16, lo f16) split:  v ~= (float)hi + (float)lo
// ---------------------------------------------------------------------------
__global__ void split_kernel(const float* __restrict__ in,
                             _Float16* __restrict__ hi,
                             _Float16* __restrict__ lo, long n)
{
    long i = (long)blockIdx.x * blockDim.x + threadIdx.x;
    if (i >= n) return;
    float v = in[i];
    _Float16 h = (_Float16)v;
    hi[i] = h;
    lo[i] = (_Float16)(v - (float)h);
}

// ---------------------------------------------------------------------------
// C[M,N] (+)= A[M,K] * W[N,K]^T, operands pre-split into f16 hi/lo.
// block = 128 threads (4 wave32), 64x64 tile, K-step 32.
// acc += Alo*Bhi + Ahi*Blo + Ahi*Bhi  (3 WMMAs, f32 accumulate).
// ---------------------------------------------------------------------------
__global__ __launch_bounds__(128)
void gemm_wmma_splitf16(const _Float16* __restrict__ Ah, const _Float16* __restrict__ Al,
                        const _Float16* __restrict__ Wh, const _Float16* __restrict__ Wl,
                        float* __restrict__ C, int M, int N, int K, int beta)
{
    __shared__ __align__(16) _Float16 As_hi[TILE_M][LDSTRIDE];
    __shared__ __align__(16) _Float16 As_lo[TILE_M][LDSTRIDE];
    __shared__ __align__(16) _Float16 Ws_hi[TILE_N][LDSTRIDE];
    __shared__ __align__(16) _Float16 Ws_lo[TILE_N][LDSTRIDE];

    const int tid    = threadIdx.x;
    const int wave   = tid >> 5;
    const int lane   = tid & 31;
    const int laneLo = lane & 15;
    const int hiHalf = lane >> 4;                 // 0: lanes 0-15, 1: lanes 16-31

    const long rowBase = (long)blockIdx.x * TILE_M;
    const long colBase = (long)blockIdx.y * TILE_N;

    v8f acc[4];
    #pragma unroll
    for (int t = 0; t < 4; ++t)
        #pragma unroll
        for (int r = 0; r < 8; ++r) acc[t][r] = 0.0f;

    for (int k0 = 0; k0 < K; k0 += TILE_K) {
        // Stage f16 tiles with 16-byte vector loads/stores.
        // 64 rows x 4 chunks of 8 halves = 256 chunks per array, 128 threads.
        #pragma unroll
        for (int it = 0; it < 2; ++it) {
            int i = tid + it * 128;
            int r = i >> 2;             // 0..63
            int c = (i & 3) * 8;        // 0,8,16,24 halves
            long ga = (rowBase + r) * (long)K + k0 + c;
            long gw = (colBase + r) * (long)K + k0 + c;
            *(v8h*)&As_hi[r][c] = *(const v8h*)&Ah[ga];
            *(v8h*)&As_lo[r][c] = *(const v8h*)&Al[ga];
            *(v8h*)&Ws_hi[r][c] = *(const v8h*)&Wh[gw];
            *(v8h*)&Ws_lo[r][c] = *(const v8h*)&Wl[gw];
            if (k0 + TILE_K < K) {
                __builtin_prefetch(&Ah[ga + TILE_K], 0, 1);
                __builtin_prefetch(&Wh[gw + TILE_K], 0, 1);
            }
        }
        __syncthreads();

        // A fragment (16x32 f16 ISA layout): lane holds row M=lane%16;
        // lanes 0-15: K = {0..7, 16..23}; lanes 16-31: K = {8..15, 24..31}.
        // Each half-run of 8 is contiguous in LDS -> one b128 load each.
        const int m    = wave * 16 + laneLo;
        const int aoff = hiHalf * 8;
        v8h a0h = *(const v8h*)&As_hi[m][aoff];
        v8h a1h = *(const v8h*)&As_hi[m][aoff + 16];
        v8h a0l = *(const v8h*)&As_lo[m][aoff];
        v8h a1l = *(const v8h*)&As_lo[m][aoff + 16];
        v16h a_hi = __builtin_shufflevector(a0h, a1h, 0,1,2,3,4,5,6,7,8,9,10,11,12,13,14,15);
        v16h a_lo = __builtin_shufflevector(a0l, a1l, 0,1,2,3,4,5,6,7,8,9,10,11,12,13,14,15);

        #pragma unroll
        for (int t = 0; t < 4; ++t) {
            // B fragment (32x16 f16 ISA layout): lane holds col N=lane%16;
            // lanes 0-15: K = {0..7, 16..23}; lanes 16-31: K = {8..15, 24..31}.
            const int n    = t * 16 + laneLo;
            const int boff = hiHalf * 8;
            v8h b0h = *(const v8h*)&Ws_hi[n][boff];
            v8h b1h = *(const v8h*)&Ws_hi[n][boff + 16];
            v8h b0l = *(const v8h*)&Ws_lo[n][boff];
            v8h b1l = *(const v8h*)&Ws_lo[n][boff + 16];
            v16h b_hi = __builtin_shufflevector(b0h, b1h, 0,1,2,3,4,5,6,7,8,9,10,11,12,13,14,15);
            v16h b_lo = __builtin_shufflevector(b0l, b1l, 0,1,2,3,4,5,6,7,8,9,10,11,12,13,14,15);

            acc[t] = __builtin_amdgcn_wmma_f32_16x16x32_f16(false, a_lo, false, b_hi,
                                                            (short)0, acc[t], false, false);
            acc[t] = __builtin_amdgcn_wmma_f32_16x16x32_f16(false, a_hi, false, b_lo,
                                                            (short)0, acc[t], false, false);
            acc[t] = __builtin_amdgcn_wmma_f32_16x16x32_f16(false, a_hi, false, b_hi,
                                                            (short)0, acc[t], false, false);
        }
        __syncthreads();
    }

    // C/D layout: VGPR r -> M = r + 8*hiHalf, N = lane%16.
    #pragma unroll
    for (int t = 0; t < 4; ++t) {
        #pragma unroll
        for (int r = 0; r < 8; ++r) {
            long row = rowBase + wave * 16 + r + hiHalf * 8;
            long col = colBase + t * 16 + laneLo;
            long idx = row * (long)N + col;
            float v = acc[t][r];
            C[idx] = beta ? (C[idx] + v) : v;
        }
    }
}

// ---------------------------------------------------------------------------
__global__ void embed_kernel(const int* __restrict__ ids,
                             const float* __restrict__ emb,
                             float* __restrict__ x)
{
    long idx = (long)blockIdx.x * blockDim.x + threadIdx.x;
    if (idx >= (long)LSEQ * DMODEL) return;
    int l = (int)(idx / DMODEL), c = (int)(idx % DMODEL);
    x[idx] = emb[(long)ids[l] * DMODEL + c];
}

__global__ __launch_bounds__(256)
void rmsnorm_kernel(const float* __restrict__ x, const float* __restrict__ w,
                    float* __restrict__ out)
{
    __shared__ float red[256];
    const int l = blockIdx.x;
    const float* xr_ = x + (long)l * DMODEL;
    float s = 0.f;
    for (int i = threadIdx.x; i < DMODEL; i += 256) { float v = xr_[i]; s += v * v; }
    red[threadIdx.x] = s;
    __syncthreads();
    for (int off = 128; off > 0; off >>= 1) {
        if (threadIdx.x < off) red[threadIdx.x] += red[threadIdx.x + off];
        __syncthreads();
    }
    float inv = rsqrtf(red[0] / (float)DMODEL + 1e-5f);
    for (int i = threadIdx.x; i < DMODEL; i += 256)
        out[(long)l * DMODEL + i] = xr_[i] * inv * w[i];
}

// Causal depthwise conv1d (DCONV=4) + SiLU; xs = first half of xr row.
__global__ void conv_silu_kernel(const float* __restrict__ xr,
                                 const float* __restrict__ cw,
                                 const float* __restrict__ cb,
                                 float* __restrict__ xc)
{
    long idx = (long)blockIdx.x * blockDim.x + threadIdx.x;
    if (idx >= (long)LSEQ * DIN) return;
    int l = (int)(idx / DIN), d = (int)(idx % DIN);
    float acc = cb[d];
    #pragma unroll
    for (int k = 0; k < 4; ++k) {
        int ll = l - 3 + k;
        if (ll >= 0) acc += xr[(long)ll * (2 * DIN) + d] * cw[d * 4 + k];
    }
    float sig = 1.f / (1.f + __expf(-acc));
    xc[idx] = acc * sig;
}

// x_dbl[l, 0..79] = xc[l,:] @ x_proj_w^T ; wave-per-output shuffle reduce.
__global__ __launch_bounds__(256)
void xproj_kernel(const float* __restrict__ xc, const float* __restrict__ Wp,
                  float* __restrict__ xdbl)
{
    __shared__ float row[DIN];
    const int l = blockIdx.x;
    for (int i = threadIdx.x; i < DIN; i += 256) row[i] = xc[(long)l * DIN + i];
    __syncthreads();
    const int wave = threadIdx.x >> 5, lane = threadIdx.x & 31;
    for (int o = wave; o < DTRANK + 2 * NST; o += 8) {
        const float* w = Wp + (long)o * DIN;
        float s = 0.f;
        for (int k = lane; k < DIN; k += 32) s += row[k] * w[k];
        #pragma unroll
        for (int off = 16; off > 0; off >>= 1) s += __shfl_down(s, off, 32);
        if (lane == 0) xdbl[(long)l * 80 + o] = s;
    }
}

// delta[l,j] = softplus(x_dbl[l,0:48] @ dt_proj_w[j,:] + dt_proj_b[j])
__global__ __launch_bounds__(256)
void dt_kernel(const float* __restrict__ xdbl, const float* __restrict__ dtW,
               const float* __restrict__ dtB, float* __restrict__ delta)
{
    __shared__ float s[DTRANK];
    const int l = blockIdx.x;
    if (threadIdx.x < DTRANK) s[threadIdx.x] = xdbl[(long)l * 80 + threadIdx.x];
    __syncthreads();
    for (int j = threadIdx.x; j < DIN; j += 256) {
        const float* w = dtW + (long)j * DTRANK;
        float acc = dtB[j];
        #pragma unroll
        for (int k = 0; k < DTRANK; ++k) acc += s[k] * w[k];
        delta[(long)l * DIN + j] = (acc > 20.f) ? acc : log1pf(__expf(acc));
    }
}

// Selective scan, fused with +u*D and *silu(res). 6 blocks x 256 ch.
__global__ __launch_bounds__(256)
void scan_kernel(const float* __restrict__ xc, const float* __restrict__ delta,
                 const float* __restrict__ xdbl, const float* __restrict__ A_log,
                 const float* __restrict__ Dv, const float* __restrict__ xr,
                 float* __restrict__ y)
{
    __shared__ float Bs[NST], Cs[NST];
    const int d = blockIdx.x * 256 + threadIdx.x;
    float Ad[NST], st[NST];
    #pragma unroll
    for (int n = 0; n < NST; ++n) {
        Ad[n] = -__expf(A_log[(long)d * NST + n]);
        st[n] = 0.f;
    }
    const float Dd = Dv[d];
    for (int l = 0; l < LSEQ; ++l) {
        if (threadIdx.x < 32) {
            // lanes 0-15 -> B (cols 48..63), lanes 16-31 -> C (cols 64..79)
            float v = xdbl[(long)l * 80 + DTRANK + threadIdx.x];
            if (threadIdx.x < 16) Bs[threadIdx.x] = v;
            else                  Cs[threadIdx.x - 16] = v;
        }
        __syncthreads();
        float dl = delta[(long)l * DIN + d];
        float u  = xc[(long)l * DIN + d];
        float du = dl * u;
        float acc = 0.f;
        #pragma unroll
        for (int n = 0; n < NST; ++n) {
            float dA = __expf(dl * Ad[n]);
            st[n] = dA * st[n] + du * Bs[n];
            acc += st[n] * Cs[n];
        }
        float res = xr[(long)l * (2 * DIN) + DIN + d];
        float sig = 1.f / (1.f + __expf(-res));
        y[(long)l * DIN + d] = (acc + u * Dd) * (res * sig);
        __syncthreads();
    }
}

// ---------------------------------------------------------------------------
static inline long roundup8(long x) { return (x + 7) & ~7L; }

extern "C" void kernel_launch(void* const* d_in, const int* in_sizes, int n_in,
                              void* d_out, int out_size, void* d_ws, size_t ws_size,
                              hipStream_t stream)
{
    const int*   ids    = (const int*)  d_in[0];
    const float* emb    = (const float*)d_in[1];
    const float* norm_w = (const float*)d_in[2];
    const float* in_w   = (const float*)d_in[3];
    const float* conv_w = (const float*)d_in[4];
    const float* conv_b = (const float*)d_in[5];
    const float* xproj_w= (const float*)d_in[6];
    const float* dt_w   = (const float*)d_in[7];
    const float* dt_b   = (const float*)d_in[8];
    const float* A_log  = (const float*)d_in[9];
    const float* Dp     = (const float*)d_in[10];
    const float* out_w  = (const float*)d_in[11];
    const float* normf  = (const float*)d_in[12];
    float* logits = (float*)d_out;

    // ---- workspace arena (units: floats; every sub-buffer 32B aligned) ----
    float* ws = (float*)d_ws;
    long off = 0;
    // persistent region
    float*    x  = ws + off;  off += roundup8((long)LSEQ * DMODEL);
    float*    h  = ws + off;  off += roundup8((long)LSEQ * DMODEL);
    _Float16* hh = (_Float16*)(ws + off);  off += roundup8((long)LSEQ * DMODEL / 2);
    _Float16* hl = (_Float16*)(ws + off);  off += roundup8((long)LSEQ * DMODEL / 2);
    // scratch arena: per-layer buffers, re-used by emb hi/lo in final phase
    long scratch0 = off;
    float*    xr    = ws + off;  off += roundup8((long)LSEQ * 2 * DIN);
    float*    xc    = ws + off;  off += roundup8((long)LSEQ * DIN);
    float*    delta = ws + off;  off += roundup8((long)LSEQ * DIN);
    float*    ybuf  = ws + off;  off += roundup8((long)LSEQ * DIN);
    float*    xdbl  = ws + off;  off += roundup8((long)LSEQ * 80);
    _Float16* yh    = (_Float16*)(ws + off);  off += roundup8((long)LSEQ * DIN / 2);
    _Float16* yl    = (_Float16*)(ws + off);  off += roundup8((long)LSEQ * DIN / 2);
    _Float16* wih   = (_Float16*)(ws + off);  off += roundup8((long)2 * DIN * DMODEL / 2);
    _Float16* wil   = (_Float16*)(ws + off);  off += roundup8((long)2 * DIN * DMODEL / 2);
    _Float16* owh   = (_Float16*)(ws + off);  off += roundup8((long)DMODEL * DIN / 2);
    _Float16* owl   = (_Float16*)(ws + off);  off += roundup8((long)DMODEL * DIN / 2);
    // final phase aliases the scratch arena
    long eoff = scratch0;
    _Float16* eh = (_Float16*)(ws + eoff);  eoff += roundup8((long)VOCAB * DMODEL / 2);
    _Float16* el = (_Float16*)(ws + eoff);

    const long nH  = (long)LSEQ * DMODEL;      // 1.57M
    const long nY  = (long)LSEQ * DIN;         // 3.15M
    const long nIW = (long)2 * DIN * DMODEL;   // 2.36M
    const long nOW = (long)DMODEL * DIN;       // 1.18M
    const long nE  = (long)VOCAB * DMODEL;     // 24.6M

    embed_kernel<<<(LSEQ * DMODEL + 255) / 256, 256, 0, stream>>>(ids, emb, x);

    for (int layer = 0; layer < NLAYER; ++layer) {
        rmsnorm_kernel<<<LSEQ, 256, 0, stream>>>(x, norm_w + layer * DMODEL, h);

        split_kernel<<<(int)((nH + 255) / 256), 256, 0, stream>>>(h, hh, hl, nH);
        split_kernel<<<(int)((nIW + 255) / 256), 256, 0, stream>>>(
            in_w + (long)layer * nIW, wih, wil, nIW);

        dim3 g1(LSEQ / TILE_M, (2 * DIN) / TILE_N);
        gemm_wmma_splitf16<<<g1, 128, 0, stream>>>(
            hh, hl, wih, wil, xr, LSEQ, 2 * DIN, DMODEL, 0);

        conv_silu_kernel<<<(LSEQ * DIN + 255) / 256, 256, 0, stream>>>(
            xr, conv_w + layer * DIN * 4, conv_b + layer * DIN, xc);

        xproj_kernel<<<LSEQ, 256, 0, stream>>>(
            xc, xproj_w + (long)layer * 80 * DIN, xdbl);

        dt_kernel<<<LSEQ, 256, 0, stream>>>(
            xdbl, dt_w + (long)layer * DIN * DTRANK, dt_b + layer * DIN, delta);

        scan_kernel<<<DIN / 256, 256, 0, stream>>>(
            xc, delta, xdbl, A_log + (long)layer * DIN * NST,
            Dp + layer * DIN, xr, ybuf);

        split_kernel<<<(int)((nY + 255) / 256), 256, 0, stream>>>(ybuf, yh, yl, nY);
        split_kernel<<<(int)((nOW + 255) / 256), 256, 0, stream>>>(
            out_w + (long)layer * nOW, owh, owl, nOW);

        dim3 g2(LSEQ / TILE_M, DMODEL / TILE_N);
        gemm_wmma_splitf16<<<g2, 128, 0, stream>>>(
            yh, yl, owh, owl, x, LSEQ, DMODEL, DIN, 1);  // residual accumulate
    }

    rmsnorm_kernel<<<LSEQ, 256, 0, stream>>>(x, normf, h);
    split_kernel<<<(int)((nH + 255) / 256), 256, 0, stream>>>(h, hh, hl, nH);
    split_kernel<<<(int)((nE + 255) / 256), 256, 0, stream>>>(emb, eh, el, nE);

    dim3 g3(LSEQ / TILE_M, VOCAB / TILE_N);
    gemm_wmma_splitf16<<<g3, 128, 0, stream>>>(
        hh, hl, eh, el, logits, LSEQ, VOCAB, DMODEL, 0);
}
